// W4A16Linear_1176821039701
// MI455X (gfx1250) — matrix operand order repaired and checked by reading, not compiled, not run on hardware
//
#include <hip/hip_runtime.h>

// ---------------------------------------------------------------------------
// W4A16 linear: out[16,11008] = x[16,4096](f16) @ dequant(W4)[11008,4096]^T
// weight_packed: int32[11008,2048], low byte = two unsigned nibbles
// (low nibble = even K, high nibble = odd K), two's-complement int4,
// per-output-channel fp16 scale.
//
// Memory-bound (90 MB weight stream vs 1.44 GFLOP; fits in 192 MB L2 so
// timed replays stream from L2). Strategy:
//  - f16 WMMA 16x16x32, M=16 = batch, one 16-col tile per wave
//  - x staged into LDS with GLOBAL_LOAD_ASYNC_TO_LDS_B128 (ASYNCcnt path,
//    no VGPR round-trip); padded rows -> conflict-free ds_load_b128 A frags
//  - weights streamed as 2x global_load_b128 per K-step per lane, nibble ->
//    fp16 via exponent-trick: ((w<<12)|w)&0x000F000F ^ 0x64086408, then one
//    v_pk_add_f16 of -1032 (lshl_or + bitop3 friendly)
//  - K split x4 across blocks (2752 waves) for bandwidth concurrency;
//    f32 partials in d_ws (written once each -> deterministic), tiny reduce
//    kernel applies per-channel scale and converts to fp16.
// ---------------------------------------------------------------------------

typedef __attribute__((ext_vector_type(16))) _Float16 v16h;
typedef __attribute__((ext_vector_type(2)))  _Float16 h2;
typedef __attribute__((ext_vector_type(8)))  float    v8f;
typedef unsigned __attribute__((ext_vector_type(4)))  u32x4;

#define IN_F    4096
#define OUT_F   11008
#define BATCH   16
#define KSPLIT  4
#define KSEG    (IN_F / KSPLIT)       // 1024 halves of K per segment
#define KSTEPS  (KSEG / 32)           // 32 WMMA K-steps per segment
#define ROWPAD  8                     // 16B pad -> conflict-free ds_load_b128
#define LDS_ROW (KSEG + ROWPAD)       // 1032 halves per staged row
#define WAVES   8                     // 256 threads; block covers 128 columns

struct APair { u32x4 lo, hi; };       // 32B == v16h
struct BPack { unsigned u[8]; };      // 32B == v16h

// Dequant one packed byte (low byte of w) -> two signed-int4 values as packed
// fp16.  (n ^ 8) - 8 sign-extends a nibble; embedding the nibble in the
// mantissa of 1024.0h and subtracting 1032.0h does convert+sub in one
// v_pk_add_f16.  ((w<<12)|w) & 0x000F000F places low nibble in half 0 and
// high nibble in half 1; bits 8..31 of w are zero (values < 256), and the
// mask kills everything else.  Shaped for v_lshl_or_b32 + v_bitop3_b32.
__device__ __forceinline__ unsigned dq_byte(unsigned w) {
    unsigned t = (((w << 12) | w) & 0x000F000Fu) ^ 0x64086408u;
    h2 v = __builtin_bit_cast(h2, t) - __builtin_bit_cast(h2, 0x64086408u);
    return __builtin_bit_cast(unsigned, v);
}

template <bool SPLIT>
__global__ __launch_bounds__(256)
void w4a16_wmma_kernel(const _Float16* __restrict__ x,
                       const int*      __restrict__ wpk,
                       const _Float16* __restrict__ scale,
                       _Float16*       __restrict__ out,
                       float*          __restrict__ ws) {
    // Only __shared__ allocation in this kernel -> lives at LDS offset 0,
    // so plain element byte offsets are valid async-to-LDS destinations.
    __shared__ _Float16 xs[BATCH * LDS_ROW];          // 33,024 bytes

    const int tid  = threadIdx.x;
    const int wave = tid >> 5;
    const int lane = tid & 31;
    const int nloc = lane & 15;       // B column / A row / C column for this lane
    const int hsel = lane >> 4;       // 0: K-low half, 1: K-high half of fragment
    const int ocol = blockIdx.x * (WAVES * 16) + wave * 16 + nloc;

    v8f acc = 0.0f;                   // 16x16 f32 accumulator tile

    const int seg0  = SPLIT ? (int)blockIdx.y : 0;
    const int nsegs = SPLIT ? 1 : KSPLIT;

    for (int s = 0; s < nsegs; ++s) {
        const int seg = seg0 + s;

        // ---- async-stage x[:, seg*KSEG ..) into padded LDS (32 KB) --------
        // GVS mode: saddr = x + seg*KSEG, vaddr = per-lane 32-bit byte offset,
        // vdst = per-lane LDS byte offset.  Tracked by ASYNCcnt.
        {
            const unsigned long long xbase =
                (unsigned long long)(const void*)(x) + (size_t)seg * KSEG * 2;
            for (int u = tid; u < BATCH * (KSEG / 8); u += 256) {
                const int row = u >> 7;                // KSEG/8 == 128 16B units
                const int c   = u & 127;
                const unsigned goff = (unsigned)(row * (IN_F * 2)    + c * 16);
                const unsigned loff = (unsigned)(row * (LDS_ROW * 2) + c * 16);
                asm volatile("global_load_async_to_lds_b128 %0, %1, %2"
                             :: "v"(loff), "v"(goff), "s"(xbase)
                             : "memory");
            }
            asm volatile("s_wait_asynccnt 0" ::: "memory");
        }
        __syncthreads();

        // ---- stream weights + WMMA ----------------------------------------
        const _Float16* abase = xs + nloc * LDS_ROW + hsel * 16;
        const int*      wbase = wpk + (size_t)ocol * (IN_F / 2)
                                    + (size_t)seg * (KSEG / 2) + hsel * 8;

#pragma unroll 8
        for (int kk = 0; kk < KSTEPS; ++kk) {
            // A fragment: row nloc, 16 consecutive K-halves [kb,kb+16)
            APair ap;
            ap.lo = *reinterpret_cast<const u32x4*>(abase + kk * 32);
            ap.hi = *reinterpret_cast<const u32x4*>(abase + kk * 32 + 8);
            v16h a = __builtin_bit_cast(v16h, ap);

            // B fragment: col nloc, 16 consecutive K values = 8 packed int32
            const u32x4* wp = reinterpret_cast<const u32x4*>(wbase + kk * 16);
            u32x4 w0 = wp[0];
            u32x4 w1 = wp[1];
            BPack bb;
            bb.u[0] = dq_byte(w0.x);  bb.u[1] = dq_byte(w0.y);
            bb.u[2] = dq_byte(w0.z);  bb.u[3] = dq_byte(w0.w);
            bb.u[4] = dq_byte(w1.x);  bb.u[5] = dq_byte(w1.y);
            bb.u[6] = dq_byte(w1.z);  bb.u[7] = dq_byte(w1.w);
            v16h b = __builtin_bit_cast(v16h, bb);

            acc = __builtin_amdgcn_wmma_f32_16x16x32_f16(
                false, a, false, b, (short)0, acc, false, false);
        }
        __syncthreads();   // safe to restage LDS / exit
    }

    // ---- epilogue ---------------------------------------------------------
    if (SPLIT) {
        // f32 partials: ws[seg][m][o], each slot written exactly once.
        const size_t base = (size_t)blockIdx.y * BATCH * OUT_F;
#pragma unroll
        for (int r = 0; r < 8; ++r)
            ws[base + (size_t)(r + hsel * 8) * OUT_F + ocol] = acc[r];
    } else {
        const float sc = (float)scale[ocol];
#pragma unroll
        for (int r = 0; r < 8; ++r)
            out[(size_t)(r + hsel * 8) * OUT_F + ocol] = (_Float16)(acc[r] * sc);
    }
}

__global__ __launch_bounds__(256)
void w4a16_reduce_kernel(const float* __restrict__ ws,
                         const _Float16* __restrict__ scale,
                         _Float16* __restrict__ out) {
    const unsigned idx = blockIdx.x * 256u + threadIdx.x;
    if (idx >= BATCH * OUT_F) return;
    const unsigned m = idx / OUT_F;
    const unsigned o = idx - m * OUT_F;
    float s = 0.0f;
#pragma unroll
    for (int k = 0; k < KSPLIT; ++k)
        s += ws[(size_t)(k * BATCH + m) * OUT_F + o];
    out[idx] = (_Float16)(s * (float)scale[o]);
}

extern "C" void kernel_launch(void* const* d_in, const int* in_sizes, int n_in,
                              void* d_out, int out_size, void* d_ws, size_t ws_size,
                              hipStream_t stream) {
    (void)in_sizes; (void)n_in; (void)out_size;
    const _Float16* x     = (const _Float16*)d_in[0];
    const int*      wpk   = (const int*)d_in[1];
    const _Float16* scale = (const _Float16*)d_in[2];
    _Float16*       out   = (_Float16*)d_out;

    const size_t need = (size_t)KSPLIT * BATCH * OUT_F * sizeof(float);
    if (ws_size >= need) {
        float* ws = (float*)d_ws;
        dim3 grid(OUT_F / (WAVES * 16), KSPLIT);     // 86 x 4
        w4a16_wmma_kernel<true><<<grid, 256, 0, stream>>>(x, wpk, scale, out, ws);
        w4a16_reduce_kernel<<<(BATCH * OUT_F + 255) / 256, 256, 0, stream>>>(ws, scale, out);
    } else {
        dim3 grid(OUT_F / (WAVES * 16), 1);          // fallback: K loop in-kernel
        w4a16_wmma_kernel<false><<<grid, 256, 0, stream>>>(x, wpk, scale, out, nullptr);
    }
}
